// HyenaDNA_54520314856279
// MI455X (gfx1250) — matrix operand order repaired and checked by reading, not compiled, not run on hardware
//
#include <hip/hip_runtime.h>
#include <hip/hip_bf16.h>

typedef __attribute__((ext_vector_type(16))) _Float16 v16h;
typedef __attribute__((ext_vector_type(8)))  float    v8f;

#define BB 4
#define SS 4096
#define DD 512
#define OO 4
#define LL 4
#define FF 2048
#define ROWS (BB * SS)

// ---------------------------------------------------------------------------
// WMMA fragment loaders (layouts per CDNA5 ISA §7.12.2)
// ---------------------------------------------------------------------------
__device__ inline v16h load_frag_a(const _Float16* __restrict__ A, int lda,
                                   int m0, int k0, int lane) {
  // A row-major (M x K). Lane holds row m0 + lane%16.
  // VGPR v holds K pair: (v<4 ? v : 4+v) + (lane>=16 ? 4 : 0)  [pair units]
  union { v16h h; unsigned int u[8]; } f;
  const unsigned int* row =
      (const unsigned int*)(A + (size_t)(m0 + (lane & 15)) * lda + k0);
  int sel = (lane & 16) ? 4 : 0;
#pragma unroll
  for (int v = 0; v < 8; ++v) {
    int p = (v < 4 ? v : 4 + v) + sel;
    f.u[v] = row[p];
  }
  return f.h;
}

__device__ inline v16h load_frag_b(const _Float16* __restrict__ Bt, int ldb,
                                   int n0, int k0, int lane) {
  // Bt is B transposed: (N x K) row-major. Lane holds column n0 + lane%16,
  // VGPR v holds K pair: (v<4 ? v : 4+v) + (lane>=16 ? 4 : 0)  [pair units]
  union { v16h h; unsigned int u[8]; } f;
  const unsigned int* col =
      (const unsigned int*)(Bt + (size_t)(n0 + (lane & 15)) * ldb + k0);
  int sel = (lane & 16) ? 4 : 0;
#pragma unroll
  for (int v = 0; v < 8; ++v) f.u[v] = col[(v < 4 ? v : 4 + v) + sel];
  return f.h;
}

__device__ inline v8f wmma_f16(v16h a, v16h b, v8f c) {
  return __builtin_amdgcn_wmma_f32_16x16x32_f16(false, a, false, b,
                                                (short)0, c, false, false);
}

// MODE 0: outH = f16(gelu(acc + bias))       (GEMM1)
// MODE 1: outF = acc + bias + res; outH=f16  (GEMM2 + residual)
// MODE 2: outF += acc + bias                 (skip accumulation)
template <int MODE>
__device__ inline void store_tile(v8f acc, int m0, int n0, int lane, int N,
                                  const float* __restrict__ bias,
                                  const float* __restrict__ res,
                                  float* __restrict__ outF,
                                  _Float16* __restrict__ outH) {
  int col  = n0 + (lane & 15);
  int row0 = m0 + ((lane & 16) ? 8 : 0);
  float bb = bias[col];
#pragma unroll
  for (int r = 0; r < 8; ++r) {
    size_t idx = (size_t)(row0 + r) * N + col;
    float xv = acc[r] + bb;
    if (MODE == 0) {
      float g = 0.5f * xv * (1.0f + erff(xv * 0.70710678118654752f));
      outH[idx] = (_Float16)g;
    } else if (MODE == 1) {
      xv += res[idx];
      outF[idx] = xv;
      outH[idx] = (_Float16)xv;
    } else {
      outF[idx] += xv;
    }
  }
}

template <int MODE>
__global__ void gemm_wmma(const _Float16* __restrict__ A,
                          const _Float16* __restrict__ Bt,
                          const float* __restrict__ bias,
                          const float* __restrict__ res,
                          float* __restrict__ outF,
                          _Float16* __restrict__ outH,
                          int M, int N, int K) {
  int wave = (int)((blockIdx.x * blockDim.x + threadIdx.x) >> 5);
  int lane = threadIdx.x & 31;
  int Nt = N >> 5;            // 32-wide tiles in N
  int Mt = M >> 5;            // 32-wide tiles in M
  if (wave >= Mt * Nt) return;            // wave-uniform: EXEC stays all-1s
  int tm = (wave / Nt) * 32;
  int tn = (wave % Nt) * 32;

  v8f acc00 = {}, acc01 = {}, acc10 = {}, acc11 = {};
  for (int k0 = 0; k0 < K; k0 += 32) {
    v16h a0 = load_frag_a(A, K, tm,      k0, lane);
    v16h a1 = load_frag_a(A, K, tm + 16, k0, lane);
    v16h b0 = load_frag_b(Bt, K, tn,      k0, lane);
    v16h b1 = load_frag_b(Bt, K, tn + 16, k0, lane);
    acc00 = wmma_f16(a0, b0, acc00);
    acc01 = wmma_f16(a0, b1, acc01);
    acc10 = wmma_f16(a1, b0, acc10);
    acc11 = wmma_f16(a1, b1, acc11);
  }
  store_tile<MODE>(acc00, tm,      tn,      lane, N, bias, res, outF, outH);
  store_tile<MODE>(acc01, tm,      tn + 16, lane, N, bias, res, outF, outH);
  store_tile<MODE>(acc10, tm + 16, tn,      lane, N, bias, res, outF, outH);
  store_tile<MODE>(acc11, tm + 16, tn + 16, lane, N, bias, res, outF, outH);
}

// ---------------------------------------------------------------------------
// Elementwise / small kernels
// ---------------------------------------------------------------------------
__global__ void embed_pe(const float* __restrict__ x,
                         const float* __restrict__ ew,
                         const float* __restrict__ eb,
                         const float* __restrict__ pe,
                         float* __restrict__ h) {
  size_t idx = (size_t)blockIdx.x * blockDim.x + threadIdx.x;
  if (idx >= (size_t)ROWS * DD) return;
  int row = (int)(idx / DD);
  int d   = (int)(idx % DD);
  int s   = row % SS;
  const float* xr = x + (size_t)row * 4;
  float v = eb[d] + xr[0] * ew[0 * DD + d] + xr[1] * ew[1 * DD + d] +
            xr[2] * ew[2 * DD + d] + xr[3] * ew[3 * DD + d];
  h[idx] = v + pe[(size_t)s * DD + d];
}

__global__ void zero_f32(float* __restrict__ p, size_t n) {
  size_t idx = (size_t)blockIdx.x * blockDim.x + threadIdx.x;
  if (idx < n) p[idx] = 0.0f;
}

// dst[c*R + r] = (f16) src[r*C + c]
__global__ void conv_transpose_f16(const float* __restrict__ src,
                                   _Float16* __restrict__ dst, int R, int C) {
  size_t idx = (size_t)blockIdx.x * blockDim.x + threadIdx.x;
  if (idx >= (size_t)R * C) return;
  int r = (int)(idx / C);
  int c = (int)(idx % C);
  dst[(size_t)c * R + r] = (_Float16)src[idx];
}

// LayerNorm over D=512, one row per block of 256 threads.
// OUTF16=0: write f32 to outF.  OUTF16=1: write f16 to outH.
template <int OUTF16>
__global__ void layernorm_k(const float* __restrict__ in,
                            const float* __restrict__ w,
                            const float* __restrict__ b,
                            float* __restrict__ outF,
                            _Float16* __restrict__ outH) {
  __shared__ float red[256];
  int row = blockIdx.x;
  int t = threadIdx.x;
  const float* r = in + (size_t)row * DD;
  float v0 = r[t], v1 = r[t + 256];

  red[t] = v0 + v1;
  __syncthreads();
  for (int off = 128; off > 0; off >>= 1) {
    if (t < off) red[t] += red[t + off];
    __syncthreads();
  }
  float mu = red[0] * (1.0f / DD);
  __syncthreads();

  float d0 = v0 - mu, d1 = v1 - mu;
  red[t] = d0 * d0 + d1 * d1;
  __syncthreads();
  for (int off = 128; off > 0; off >>= 1) {
    if (t < off) red[t] += red[t + off];
    __syncthreads();
  }
  float rstd = rsqrtf(red[0] * (1.0f / DD) + 1e-5f);

  float o0 = d0 * rstd * w[t] + b[t];
  float o1 = d1 * rstd * w[t + 256] + b[t + 256];
  if (OUTF16) {
    _Float16* oh = outH + (size_t)row * DD;
    oh[t] = (_Float16)o0;
    oh[t + 256] = (_Float16)o1;
  } else {
    float* of = outF + (size_t)row * DD;
    of[t] = o0;
    of[t + 256] = o1;
  }
}

// proj[s*B*O + b*O + o] = sum_d hn[b,s,d] * P[o,d] ; one (b,s) row per block.
__global__ void hyena_proj(const float* __restrict__ hn,
                           const float* __restrict__ P,
                           float* __restrict__ proj) {
  __shared__ float red[256];
  int row = blockIdx.x;                 // row = b*S + s
  int b = row / SS, s = row % SS;
  int t = threadIdx.x;
  const float* r = hn + (size_t)row * DD;
  float v0 = r[t], v1 = r[t + 256];
#pragma unroll
  for (int o = 0; o < OO; ++o) {
    red[t] = v0 * P[o * DD + t] + v1 * P[o * DD + t + 256];
    __syncthreads();
    for (int off = 128; off > 0; off >>= 1) {
      if (t < off) red[t] += red[t + off];
      __syncthreads();
    }
    if (t == 0) proj[(size_t)s * BB * OO + b * OO + o] = red[0];
    __syncthreads();
  }
}

// Causal linear recurrence, summed over orders, added in-place to h (residual).
// grid = B blocks, 512 threads (one per d). proj staged through LDS.
__global__ void hyena_scan(const float* __restrict__ proj,
                           const float* __restrict__ Cf,
                           float* __restrict__ h) {
  __shared__ float pbuf[128 * OO];
  int b = blockIdx.x;
  int d = threadIdx.x;                  // 0..511
  float c[OO], carry[OO];
#pragma unroll
  for (int o = 0; o < OO; ++o) {
    c[o] = expf(Cf[o * DD + d]);
    carry[o] = 0.0f;
  }
  float* hb = h + (size_t)b * SS * DD;
  for (int t0 = 0; t0 < SS; t0 += 128) {
    int i = threadIdx.x;                // 512 floats = 128 steps * 4 orders
    pbuf[i] = proj[(size_t)(t0 + (i >> 2)) * BB * OO + b * OO + (i & 3)];
    __syncthreads();
    for (int dt = 0; dt < 128; ++dt) {
      const float* p = &pbuf[dt * OO];
      float sum = 0.0f;
#pragma unroll
      for (int o = 0; o < OO; ++o) {
        carry[o] = carry[o] * c[o] + p[o];
        sum += carry[o];
      }
      size_t idx = (size_t)(t0 + dt) * DD + d;
      hb[idx] = hb[idx] + sum;          // += residual semantics
    }
    __syncthreads();
  }
}

// Final projection (D->3) + softmax; one wave per row.
__global__ void out_softmax(const float* __restrict__ comb,
                            const float* __restrict__ ow,
                            const float* __restrict__ ob,
                            float* __restrict__ out) {
  int wid = (int)((blockIdx.x * blockDim.x + threadIdx.x) >> 5);
  int lane = threadIdx.x & 31;
  if (wid >= ROWS) return;
  const float* r = comb + (size_t)wid * DD;
  float a0 = 0.0f, a1 = 0.0f, a2 = 0.0f;
  for (int d = lane; d < DD; d += 32) {
    float v = r[d];
    a0 += v * ow[d * 3 + 0];
    a1 += v * ow[d * 3 + 1];
    a2 += v * ow[d * 3 + 2];
  }
  for (int off = 16; off > 0; off >>= 1) {
    a0 += __shfl_xor(a0, off, 32);
    a1 += __shfl_xor(a1, off, 32);
    a2 += __shfl_xor(a2, off, 32);
  }
  if (lane == 0) {
    float l0 = a0 + ob[0], l1 = a1 + ob[1], l2 = a2 + ob[2];
    float m = fmaxf(l0, fmaxf(l1, l2));
    float e0 = expf(l0 - m), e1 = expf(l1 - m), e2 = expf(l2 - m);
    float inv = 1.0f / (e0 + e1 + e2);
    out[(size_t)wid * 3 + 0] = e0 * inv;
    out[(size_t)wid * 3 + 1] = e1 * inv;
    out[(size_t)wid * 3 + 2] = e2 * inv;
  }
}

// ---------------------------------------------------------------------------
// Host launcher
// ---------------------------------------------------------------------------
static inline size_t align_up(size_t v) { return (v + 255) & ~(size_t)255; }

extern "C" void kernel_launch(void* const* d_in, const int* in_sizes, int n_in,
                              void* d_out, int out_size, void* d_ws, size_t ws_size,
                              hipStream_t stream) {
  const float* x        = (const float*)d_in[0];
  const float* embed_w  = (const float*)d_in[1];
  const float* embed_b  = (const float*)d_in[2];
  const float* pe       = (const float*)d_in[3];
  const float* norm1_w  = (const float*)d_in[4];
  const float* norm1_b  = (const float*)d_in[5];
  const float* norm2_w  = (const float*)d_in[6];
  const float* norm2_b  = (const float*)d_in[7];
  const float* hy_proj  = (const float*)d_in[8];
  const float* hy_coef  = (const float*)d_in[9];
  const float* ffn_w1   = (const float*)d_in[10];
  const float* ffn_b1   = (const float*)d_in[11];
  const float* ffn_w2   = (const float*)d_in[12];
  const float* ffn_b2   = (const float*)d_in[13];
  const float* skip_w   = (const float*)d_in[14];
  const float* skip_b   = (const float*)d_in[15];
  const float* out_w    = (const float*)d_in[16];
  const float* out_b    = (const float*)d_in[17];
  float* out = (float*)d_out;

  // workspace carve
  char* ws = (char*)d_ws;
  size_t off = 0;
  float* h    = (float*)(ws + off); off = align_up(off + (size_t)ROWS * DD * 4);
  float* hn   = (float*)(ws + off); off = align_up(off + (size_t)ROWS * DD * 4);
  float* comb = (float*)(ws + off); off = align_up(off + (size_t)ROWS * DD * 4);
  float* proj = (float*)(ws + off); off = align_up(off + (size_t)SS * BB * OO * 4);
  _Float16* hn16 = (_Float16*)(ws + off); off = align_up(off + (size_t)ROWS * DD * 2);
  _Float16* h16  = (_Float16*)(ws + off); off = align_up(off + (size_t)ROWS * DD * 2);
  _Float16* g16  = (_Float16*)(ws + off); off = align_up(off + (size_t)ROWS * FF * 2);
  _Float16* w1t  = (_Float16*)(ws + off); off = align_up(off + (size_t)FF * DD * 2);
  _Float16* w2t  = (_Float16*)(ws + off); off = align_up(off + (size_t)DD * FF * 2);
  _Float16* st   = (_Float16*)(ws + off); off = align_up(off + (size_t)DD * DD * 2);

  const int T = 256;
  size_t nHD = (size_t)ROWS * DD;

  embed_pe<<<dim3((unsigned)((nHD + T - 1) / T)), dim3(T), 0, stream>>>(
      x, embed_w, embed_b, pe, h);
  zero_f32<<<dim3((unsigned)((nHD + T - 1) / T)), dim3(T), 0, stream>>>(comb, nHD);

  for (int l = 0; l < LL; ++l) {
    // weight convert + transpose to f16 (N x K) layout
    conv_transpose_f16<<<dim3((unsigned)(((size_t)DD * FF + T - 1) / T)), dim3(T), 0, stream>>>(
        ffn_w1 + (size_t)l * DD * FF, w1t, DD, FF);          // (D,F) -> (F,D)
    conv_transpose_f16<<<dim3((unsigned)(((size_t)FF * DD + T - 1) / T)), dim3(T), 0, stream>>>(
        ffn_w2 + (size_t)l * FF * DD, w2t, FF, DD);          // (F,D) -> (D,F)
    conv_transpose_f16<<<dim3((unsigned)(((size_t)DD * DD + T - 1) / T)), dim3(T), 0, stream>>>(
        skip_w + (size_t)l * DD * DD, st, DD, DD);           // (D,D) -> (D,D)^T

    // LN1 -> hn (f32)
    layernorm_k<0><<<dim3(ROWS), dim3(256), 0, stream>>>(
        h, norm1_w + (size_t)l * DD, norm1_b + (size_t)l * DD, hn, nullptr);

    // hyena: proj then causal scan (adds into h, i.e. h = hyena(hn) + h)
    hyena_proj<<<dim3(ROWS), dim3(256), 0, stream>>>(hn, hy_proj + (size_t)l * OO * DD, proj);
    hyena_scan<<<dim3(BB), dim3(DD), 0, stream>>>(proj, hy_coef + (size_t)l * OO * DD, h);

    // LN2 -> hn16 (f16)
    layernorm_k<1><<<dim3(ROWS), dim3(256), 0, stream>>>(
        h, norm2_w + (size_t)l * DD, norm2_b + (size_t)l * DD, nullptr, hn16);

    // GEMM1: (ROWS x D) x (D x F) + b1, GELU -> g16
    {
      int waves = (ROWS / 32) * (FF / 32);
      gemm_wmma<0><<<dim3((waves + 7) / 8), dim3(256), 0, stream>>>(
          hn16, w1t, ffn_b1 + (size_t)l * FF, nullptr, nullptr, g16, ROWS, FF, DD);
    }
    // GEMM2: (ROWS x F) x (F x D) + b2 + residual(h) -> h (f32) and h16
    {
      int waves = (ROWS / 32) * (DD / 32);
      gemm_wmma<1><<<dim3((waves + 7) / 8), dim3(256), 0, stream>>>(
          g16, w2t, ffn_b2 + (size_t)l * DD, h, h, h16, ROWS, DD, FF);
    }
    // skip: comb += h @ skip_w + skip_b
    {
      int waves = (ROWS / 32) * (DD / 32);
      gemm_wmma<2><<<dim3((waves + 7) / 8), dim3(256), 0, stream>>>(
          h16, st, skip_b + (size_t)l * DD, nullptr, comb, nullptr, ROWS, DD, DD);
    }
  }

  // final projection + softmax, one wave per row
  out_softmax<<<dim3((ROWS * 32 + T - 1) / T), dim3(T), 0, stream>>>(comb, out_w, out_b, out);
}
